// GraphAttentionV2Layer_83459804496007
// MI455X (gfx1250) — compile-verified, matrix operand order checked by reading, not run on hardware
//
#include <hip/hip_runtime.h>

typedef __attribute__((ext_vector_type(16))) _Float16 v16h;
typedef __attribute__((ext_vector_type(8)))  _Float16 v8h;
typedef __attribute__((ext_vector_type(8)))  float    v8f;

// Problem constants (from reference)
constexpr int Bc  = 4;     // batch
constexpr int Nn  = 512;   // nodes
constexpr int INF = 128;   // in_features
constexpr int Hh  = 8;     // heads
constexpr int Ff  = 16;    // hidden per head
constexpr int HF  = 128;   // H*F
constexpr float SLOPE = 0.2f;

// k-index inside a 16x32 f16 WMMA operand for element e (0..15) of a lane:
//   lane-half 0: VGPR0..3 -> K 0..7, VGPR4..7 -> K 16..23
//   lane-half 1: VGPR0..3 -> K 8..15, VGPR4..7 -> K 24..31
__device__ __forceinline__ int wmma_k(int e, int khalf) {
    return ((e < 8) ? 0 : 16) + khalf * 8 + (e & 7);
}

// ---------------------------------------------------------------------------
// Kernel 1: g_l = h @ W_l^T, g_r = h @ W_r^T   (f32 in, WMMA f16->f32)
// One wave (32 threads) per 16x16 output tile; A (h tile) shared between the
// two accumulators. Also writes g_r transposed as f16 [B,H,F,N] for kernel 3.
// ---------------------------------------------------------------------------
__global__ __launch_bounds__(32) void gat_proj_kernel(
    const float* __restrict__ h,
    const float* __restrict__ Wl,
    const float* __restrict__ Wr,
    float* __restrict__ gl,          // [B*N, HF] f32
    float* __restrict__ gr,          // [B*N, HF] f32
    _Float16* __restrict__ grT)      // [B,H,F,N] f16
{
    const int lane  = threadIdx.x;       // 0..31
    const int rc    = lane & 15;         // A row / B col within tile
    const int khalf = lane >> 4;
    const int mbase = blockIdx.x * 16;   // global row (b*N+j), 128 tiles
    const int nbase = blockIdx.y * 16;   // output feature, 8 tiles

    const float* hrow  = h  + (size_t)(mbase + rc) * INF;
    const float* wlrow = Wl + (size_t)(nbase + rc) * INF;   // B[k,n] = W[n,k]
    const float* wrrow = Wr + (size_t)(nbase + rc) * INF;

    v8f accL = {};
    v8f accR = {};
    for (int k0 = 0; k0 < INF; k0 += 32) {
        v16h a, bl, br;
#pragma unroll
        for (int e = 0; e < 16; ++e) {
            const int k = k0 + wmma_k(e, khalf);
            a[e]  = (_Float16)hrow[k];
            bl[e] = (_Float16)wlrow[k];
            br[e] = (_Float16)wrrow[k];
        }
        accL = __builtin_amdgcn_wmma_f32_16x16x32_f16(
                   false, a, false, bl, (short)0, accL, false, false);
        accR = __builtin_amdgcn_wmma_f32_16x16x32_f16(
                   false, a, false, br, (short)0, accR, false, false);
    }

    // D layout: VGPR r -> M = r + 8*khalf, N = lane&15
#pragma unroll
    for (int r = 0; r < 8; ++r) {
        const int m = mbase + r + 8 * khalf;        // global row = b*N + j
        const int n = nbase + rc;                   // feature = hh*16 + f
        gl[(size_t)m * HF + n] = accL[r];
        gr[(size_t)m * HF + n] = accR[r];
        const int b  = m >> 9;                      // / Nn
        const int j  = m & (Nn - 1);
        const int hh = n >> 4;
        const int f  = n & 15;
        grT[(((size_t)b * Hh + hh) * Ff + f) * Nn + j] = (_Float16)accR[r];
    }
}

// ---------------------------------------------------------------------------
// Kernel 2: e[b,i,j,h] = sum_f w[f]*lrelu(g_r[b,i,h,f] + g_l[b,j,h,f]),
// mask by adj, softmax over j. One block per (b,i); wave w handles head w.
// Each lane keeps 16 row entries in registers; wave32 shuffle reductions.
// Writes a as f16 [B,H,N,N] (WMMA A-matrix layout friendly).
// ---------------------------------------------------------------------------
__global__ __launch_bounds__(256) void gat_attn_softmax_kernel(
    const float* __restrict__ adj,     // [B,N,N]
    const float* __restrict__ gl,      // [B*N, HF]
    const float* __restrict__ gr,      // [B*N, HF]
    const float* __restrict__ w_attn,  // [F]
    _Float16* __restrict__ a_out)      // [B,H,N,N] f16
{
    const int lane = threadIdx.x & 31;
    const int hh   = threadIdx.x >> 5;           // wave id == head
    const int b    = blockIdx.x >> 9;            // / Nn
    const int i    = blockIdx.x & (Nn - 1);

    float gri[16], w[16];
    const float* grp = gr + ((size_t)b * Nn + i) * HF + hh * Ff;
#pragma unroll
    for (int f = 0; f < 16; ++f) {
        gri[f] = grp[f];
        w[f]   = w_attn[f];
    }

    const float* adjrow = adj + ((size_t)b * Nn + i) * Nn;
    const float* glb    = gl + (size_t)b * Nn * HF + hh * Ff;

    float ev[16];
    float mx = -INFINITY;
#pragma unroll
    for (int t = 0; t < 16; ++t) {
        const int j = t * 32 + lane;
        const float* glj = glb + (size_t)j * HF;
        float dot = 0.f;
#pragma unroll
        for (int f = 0; f < 16; ++f) {
            float s = gri[f] + glj[f];
            s = (s > 0.f) ? s : SLOPE * s;        // leaky_relu inside reduction
            dot = fmaf(s, w[f], dot);
        }
        const float av = adjrow[j];
        const float e  = (av != 0.f) ? dot : -INFINITY;  // mask (adj is 0/1)
        ev[t] = e;
        mx = fmaxf(mx, e);
    }
    // wave32 max reduction
#pragma unroll
    for (int off = 16; off >= 1; off >>= 1)
        mx = fmaxf(mx, __shfl_xor(mx, off, 32));

    float sum = 0.f;
#pragma unroll
    for (int t = 0; t < 16; ++t) {
        const float x = (ev[t] == -INFINITY) ? 0.f : __expf(ev[t] - mx);
        ev[t] = x;
        sum += x;
    }
#pragma unroll
    for (int off = 16; off >= 1; off >>= 1)
        sum += __shfl_xor(sum, off, 32);
    const float inv = 1.f / sum;   // self-loops guarantee sum > 0

    _Float16* arow = a_out + (((size_t)b * Hh + hh) * Nn + i) * Nn;
#pragma unroll
    for (int t = 0; t < 16; ++t)
        arow[t * 32 + lane] = (_Float16)(ev[t] * inv);
}

// ---------------------------------------------------------------------------
// Kernel 3: attn_res[b,i,h,:] = sum_j a[b,h,i,j] * g_r[b,j,h,:] via WMMA,
// fused final leaky_relu, f32 store to out [B,N,H*F].
// One wave per 16x16 (i x f) tile per (b,h); K loop over j (16 WMMA steps).
// ---------------------------------------------------------------------------
__global__ __launch_bounds__(32) void gat_av_kernel(
    const _Float16* __restrict__ a,    // [B,H,N,N]
    const _Float16* __restrict__ grT,  // [B,H,F,N]
    float* __restrict__ out)           // [B,N,HF]
{
    const int lane  = threadIdx.x;
    const int rc    = lane & 15;
    const int khalf = lane >> 4;
    const int ibase = blockIdx.x * 16;          // 32 i-tiles
    const int b     = blockIdx.y >> 3;          // grid.y = B*H
    const int hh    = blockIdx.y & 7;

    const _Float16* arow = a   + (((size_t)b * Hh + hh) * Nn + (ibase + rc)) * Nn; // A row i
    const _Float16* bcol = grT + (((size_t)b * Hh + hh) * Ff + rc) * Nn;           // B col f

    v8f acc = {};
    for (int k0 = 0; k0 < Nn; k0 += 32) {
        // Each lane's 16 operand halves are two contiguous 8-half (128-bit) chunks.
        const v8h a0 = *(const v8h*)(arow + k0 + khalf * 8);
        const v8h a1 = *(const v8h*)(arow + k0 + 16 + khalf * 8);
        const v8h b0 = *(const v8h*)(bcol + k0 + khalf * 8);
        const v8h b1 = *(const v8h*)(bcol + k0 + 16 + khalf * 8);
        const v16h av = __builtin_shufflevector(a0, a1, 0, 1, 2, 3, 4, 5, 6, 7,
                                                8, 9, 10, 11, 12, 13, 14, 15);
        const v16h bv = __builtin_shufflevector(b0, b1, 0, 1, 2, 3, 4, 5, 6, 7,
                                                8, 9, 10, 11, 12, 13, 14, 15);
        acc = __builtin_amdgcn_wmma_f32_16x16x32_f16(
                  false, av, false, bv, (short)0, acc, false, false);
    }

#pragma unroll
    for (int r = 0; r < 8; ++r) {
        const int m = ibase + r + 8 * khalf;    // node i
        const int n = rc;                       // f
        float x = acc[r];
        x = (x > 0.f) ? x : SLOPE * x;          // final leaky_relu
        out[((size_t)b * Nn + m) * HF + hh * Ff + n] = x;
    }
}

// ---------------------------------------------------------------------------
extern "C" void kernel_launch(void* const* d_in, const int* in_sizes, int n_in,
                              void* d_out, int out_size, void* d_ws, size_t ws_size,
                              hipStream_t stream) {
    (void)in_sizes; (void)n_in; (void)out_size; (void)ws_size;
    const float* h      = (const float*)d_in[0];   // [B,N,IN_F]
    const float* adj    = (const float*)d_in[1];   // [B,N,N]
    const float* Wl     = (const float*)d_in[2];   // [HF, IN_F]
    const float* Wr     = (const float*)d_in[3];   // [HF, IN_F]
    const float* w_attn = (const float*)d_in[4];   // [F]
    float* out          = (float*)d_out;           // [B,N,HF]

    // Workspace layout (bytes):
    //   gl  f32 [B*N,HF]   @ 0         (1,048,576)
    //   gr  f32 [B*N,HF]   @ 1<<20     (1,048,576)
    //   grT f16 [B,H,F,N]  @ 2<<20     (  524,288)
    //   a   f16 [B,H,N,N]  @ 2<<20 + 1<<19  (16,777,216)   total ~18.5 MB
    char* ws = (char*)d_ws;
    float*    gl  = (float*)(ws);
    float*    gr  = (float*)(ws + (size_t)(1u << 20));
    _Float16* grT = (_Float16*)(ws + (size_t)(2u << 20));
    _Float16* aat = (_Float16*)(ws + (size_t)(2u << 20) + (size_t)(1u << 19));

    // 1) projections (both W_l and W_r, shared A loads)
    gat_proj_kernel<<<dim3((Bc * Nn) / 16, HF / 16), dim3(32), 0, stream>>>(
        h, Wl, Wr, gl, gr, grT);

    // 2) pairwise scores + mask + softmax over neighbors
    gat_attn_softmax_kernel<<<dim3(Bc * Nn), dim3(256), 0, stream>>>(
        adj, gl, gr, w_attn, aat);

    // 3) attention-weighted aggregation + final leaky_relu
    gat_av_kernel<<<dim3(Nn / 16, Bc * Hh), dim3(32), 0, stream>>>(
        aat, grT, out);
}
